// FuzzyAttentionHead_53188874993995
// MI455X (gfx1250) — matrix operand order repaired and checked by reading, not compiled, not run on hardware
//
#include <hip/hip_runtime.h>

typedef __bf16 bf16;
typedef __attribute__((ext_vector_type(16))) __bf16 v16bf;
typedef __attribute__((ext_vector_type(8)))  float  v8f;
typedef __attribute__((ext_vector_type(4)))  int    v4i;

#define B_   8
#define S_   2048
#define DM   1024
#define DK   64
#define DAUG 96        // 64 features + 3 membership slots padded to 96 (3 x K=32 WMMA chunks)

#ifndef __has_builtin
#define __has_builtin(x) 0
#endif

#if __has_builtin(__builtin_amdgcn_global_load_async_to_lds_b128)
#define HAVE_ASYNC 1
#else
#define HAVE_ASYNC 0
#endif

#if HAVE_ASYNC
# if __has_builtin(__builtin_amdgcn_s_wait_asynccnt)
#  define WAIT_ASYNC(n) __builtin_amdgcn_s_wait_asynccnt(n)
# else
#  define WAIT_ASYNC(n) asm volatile("s_wait_asynccnt %0" ::"n"(n))
# endif
#else
# define WAIT_ASYNC(n) ((void)0)
#endif

typedef __attribute__((address_space(1))) v4i gv4i;   // global b128 chunk
typedef __attribute__((address_space(3))) v4i lv4i;   // LDS b128 chunk

// one 16-byte async (or sync-fallback) copy global -> LDS
static __device__ inline void async_cp16(const bf16* g, bf16* l) {
#if HAVE_ASYNC
  __builtin_amdgcn_global_load_async_to_lds_b128((gv4i*)g, (lv4i*)l, 0, 0);
#else
  *(uint4*)l = *(const uint4*)g;
#endif
}

// ---------- fragment helpers (layouts per CDNA5 ISA 7.12.2) ----------

// 16 contiguous bf16 -> v16bf (B-operand: col = lane&15, K-half chosen by lane>=16)
__device__ inline v16bf load_b16x16(const bf16* p) {
  union { uint4 u[2]; v16bf v; } r;
  const uint4* q = (const uint4*)p;
  r.u[0] = q[0]; r.u[1] = q[1];
  return r.v;
}

// two contiguous 8-element chunks -> v16bf (A-operand: K {j0..j0+7} and {j0+8..j0+15}, j0 = 16*half)
__device__ inline v16bf load_b16_2x8(const bf16* p0, const bf16* p1) {
  union { uint4 u[2]; v16bf v; } r;
  r.u[0] = *(const uint4*)p0;
  r.u[1] = *(const uint4*)p1;
  return r.v;
}

// 8+8 floats -> v16bf
__device__ inline v16bf cvt16(const float* a, const float* b) {
  union { v16bf v; bf16 h[16]; } r;
#pragma unroll
  for (int i = 0; i < 8; ++i) { r.h[i] = (bf16)a[i]; r.h[8 + i] = (bf16)b[i]; }
  return r.v;
}

// ---------- kernel 1: projection GEMM  Out[row][j] = sum_d X[row][d] * W[j][d] ----------
__global__ __launch_bounds__(128) void proj_kernel(const float* __restrict__ X,
                                                   const float* __restrict__ W,
                                                   bf16* __restrict__ Out,
                                                   int transposedV) {
  const int wave = threadIdx.x >> 5, lane = threadIdx.x & 31;
  const int row16 = lane & 15, half = lane >> 4;
  const int rowBase = blockIdx.x * 64 + wave * 16;

  v8f acc[4];
#pragma unroll
  for (int t = 0; t < 4; ++t) acc[t] = {};

  for (int kc = 0; kc < DM; kc += 32) {
    const float* ap = X + (size_t)(rowBase + row16) * DM + kc + half * 16;
    float a0[8], a1[8];
    *(float4*)&a0[0] = *(const float4*)(ap);
    *(float4*)&a0[4] = *(const float4*)(ap + 4);
    *(float4*)&a1[0] = *(const float4*)(ap + 8);
    *(float4*)&a1[4] = *(const float4*)(ap + 12);
    v16bf a = cvt16(a0, a1);
#pragma unroll
    for (int t = 0; t < 4; ++t) {
      const float* bp = W + (size_t)(t * 16 + row16) * DM + kc + half * 16;
      float b0[8], b1[8];
      *(float4*)&b0[0] = *(const float4*)(bp);
      *(float4*)&b0[4] = *(const float4*)(bp + 4);
      *(float4*)&b1[0] = *(const float4*)(bp + 8);
      *(float4*)&b1[4] = *(const float4*)(bp + 12);
      v16bf b = cvt16(b0, b1);
      acc[t] = __builtin_amdgcn_wmma_f32_16x16x32_bf16(false, a, false, b,
                                                       (short)0, acc[t], false, false);
    }
  }

  if (!transposedV) {
#pragma unroll
    for (int t = 0; t < 4; ++t)
#pragma unroll
      for (int i = 0; i < 8; ++i)
        Out[(size_t)(rowBase + half * 8 + i) * DAUG + t * 16 + row16] = (bf16)acc[t][i];
  } else {
    const int b = rowBase / S_;
    const int s0 = (rowBase % S_) + half * 8;
#pragma unroll
    for (int t = 0; t < 4; ++t) {
      const int d = t * 16 + row16;
      union { bf16 h[8]; uint4 u; } pk;
#pragma unroll
      for (int i = 0; i < 8; ++i) pk.h[i] = (bf16)acc[t][i];
      *(uint4*)(Out + (size_t)(b * DK + d) * S_ + s0) = pk.u;
    }
  }
}

// ---------- kernel 2: gaussian memberships into augmented columns ----------
__global__ void memb_kernel(bf16* __restrict__ Qa, bf16* __restrict__ Ka,
                            const float* __restrict__ cq, const float* __restrict__ sq,
                            const float* __restrict__ ck, const float* __restrict__ sk,
                            const float* __restrict__ wp) {
  const int tid = blockIdx.x * blockDim.x + threadIdx.x;
  const int tensor = tid >> 14;
  const int row = tid & 16383;
  const float fw = 1.f / (1.f + __expf(-wp[0]));
  const float ratio = (8.f * fw) / (3.f * (1.f - fw));   // beta/alpha folded into mq
  bf16* rp = (tensor == 0 ? Qa : Ka) + (size_t)row * DAUG;
  const float* C  = (tensor == 0) ? cq : ck;
  const float* Sg = (tensor == 0) ? sq : sk;
  const float scale = (tensor == 0) ? ratio : 1.f;

  float x[DK];
#pragma unroll
  for (int d = 0; d < DK; ++d) x[d] = (float)rp[d];
#pragma unroll
  for (int n = 0; n < 3; ++n) {
    float sum = 0.f;
#pragma unroll
    for (int d = 0; d < DK; ++d) {
      float diff = (x[d] - C[n * DK + d]) / fabsf(Sg[n * DK + d]);
      sum += __expf(-0.5f * diff * diff);
    }
    rp[DK + n] = (bf16)(sum * (scale / (float)DK));
  }
#pragma unroll
  for (int j = DK + 3; j < DAUG; ++j) rp[j] = (bf16)0.f;
}

// ---------- kernel 3: fused flash attention with async double-buffered LDS tiles ----------
// 4 waves/block share each 32-key tile: K' (32x96) + Vt (64x32) staged via
// GLOBAL_LOAD_ASYNC_TO_LDS_B128 (ASYNCcnt), double-buffered; prefetch tile t+1 under tile t.
__global__ __launch_bounds__(128) void attn_kernel(const bf16* __restrict__ Qa,
                                                   const bf16* __restrict__ Ka,
                                                   const bf16* __restrict__ Vt,
                                                   float* __restrict__ out,
                                                   const float* __restrict__ wp,
                                                   const float* __restrict__ tp) {
  __shared__ bf16 sK[2][32 * DAUG];   // 2 x 6 KB
  __shared__ bf16 sV[2][DK * 32];     // 2 x 4 KB
  __shared__ bf16 sP[4][16 * 32];     // per-wave P bounce, 4 KB

  const int tid = threadIdx.x;
  const int wave = tid >> 5, lane = tid & 31;
  const int row16 = lane & 15, half = lane >> 4;
  const int qRow = blockIdx.x * 64 + wave * 16;
  const int b = qRow / S_;
  const bf16* KaB = Ka + (size_t)b * S_ * DAUG;
  const bf16* VtB = Vt + (size_t)b * DK * S_;

  const float fw = 1.f / (1.f + __expf(-(*wp)));
  const float alpha = (1.f - fw) / (8.f * fabsf(*tp));

  // Q' A-fragments, resident for whole pass
  v16bf qa[3];
#pragma unroll
  for (int c = 0; c < 3; ++c) {
    const bf16* p = Qa + (size_t)(qRow + row16) * DAUG + c * 32 + half * 16;
    qa[c] = load_b16_2x8(p, p + 8);
  }

  v8f O[4];
#pragma unroll
  for (int t = 0; t < 4; ++t) O[t] = {};
  float m[8], l[8];
#pragma unroll
  for (int i = 0; i < 8; ++i) { m[i] = -1e30f; l[i] = 0.f; }

  bf16* myP = sP[wave];

  // cooperative tile stage: 5 x 16B async copies per thread
  auto stage = [&](int sBase, int buf) {
#pragma unroll
    for (int r = 0; r < 3; ++r) {                 // K': 32 rows x 12 chunks
      int c = tid + r * 128;
      int row = c / 12, col = (c % 12) * 8;
      async_cp16(KaB + (size_t)(sBase + row) * DAUG + col, &sK[buf][row * DAUG + col]);
    }
#pragma unroll
    for (int r = 0; r < 2; ++r) {                 // Vt: 64 rows x 4 chunks
      int c = tid + r * 128;
      int d = c / 4, off = (c % 4) * 8;
      async_cp16(VtB + (size_t)d * S_ + sBase + off, &sV[buf][d * 32 + off]);
    }
  };

  const int NT = S_ / 32;
  stage(0, 0);

  for (int it = 0; it < NT; ++it) {
    if (it + 1 < NT) {
      stage((it + 1) * 32, (it + 1) & 1);
      WAIT_ASYNC(5);            // tile `it`'s 5 copies done (in-order completion)
    } else {
      WAIT_ASYNC(0);
    }
    __syncthreads();            // all waves' copies for tile `it` visible

    const bf16* Kb = sK[it & 1];
    const bf16* Vb = sV[it & 1];

    v8f s0 = {}, s1 = {};
#pragma unroll
    for (int c = 0; c < 3; ++c) {
      s0 = __builtin_amdgcn_wmma_f32_16x16x32_bf16(
          false, qa[c], false, load_b16x16(Kb + (size_t)row16 * DAUG + c * 32 + half * 16),
          (short)0, s0, false, false);
      s1 = __builtin_amdgcn_wmma_f32_16x16x32_bf16(
          false, qa[c], false, load_b16x16(Kb + (size_t)(16 + row16) * DAUG + c * 32 + half * 16),
          (short)0, s1, false, false);
    }

    // online softmax; reductions stay inside 16-lane halves (rows 0-7 vs 8-15)
#pragma unroll
    for (int i = 0; i < 8; ++i) {
      float x0 = alpha * s0[i], x1 = alpha * s1[i];
      float t = fmaxf(x0, x1);
#pragma unroll
      for (int off = 8; off > 0; off >>= 1) t = fmaxf(t, __shfl_xor(t, off, 32));
      float mnew = fmaxf(m[i], t);
      float sc = __expf(m[i] - mnew);
      m[i] = mnew;
      float p0 = __expf(x0 - mnew), p1 = __expf(x1 - mnew);
      float rs = p0 + p1;
#pragma unroll
      for (int off = 8; off > 0; off >>= 1) rs += __shfl_xor(rs, off, 32);
      l[i] = l[i] * sc + rs;
#pragma unroll
      for (int t4 = 0; t4 < 4; ++t4) O[t4][i] *= sc;
      myP[(half * 8 + i) * 32 + row16]      = (bf16)p0;
      myP[(half * 8 + i) * 32 + 16 + row16] = (bf16)p1;
    }

    // P (16x32) C-layout -> A-layout through per-wave LDS region
    const bf16* pr = myP + row16 * 32 + half * 16;
    v16bf pf = load_b16_2x8(pr, pr + 8);
#pragma unroll
    for (int t4 = 0; t4 < 4; ++t4) {
      O[t4] = __builtin_amdgcn_wmma_f32_16x16x32_bf16(
          false, pf, false, load_b16x16(Vb + (size_t)(t4 * 16 + row16) * 32 + half * 16),
          (short)0, O[t4], false, false);
    }

    __syncthreads();            // done reading buf (it&1) before it is re-staged
  }

#pragma unroll
  for (int t4 = 0; t4 < 4; ++t4)
#pragma unroll
    for (int i = 0; i < 8; ++i)
      out[(size_t)(qRow + half * 8 + i) * DK + t4 * 16 + row16] = O[t4][i] / l[i];
}

extern "C" void kernel_launch(void* const* d_in, const int* in_sizes, int n_in,
                              void* d_out, int out_size, void* d_ws, size_t ws_size,
                              hipStream_t stream) {
  const float* query = (const float*)d_in[0];
  const float* key   = (const float*)d_in[1];
  const float* value = (const float*)d_in[2];
  const float* Wq    = (const float*)d_in[3];
  const float* Wk    = (const float*)d_in[4];
  const float* Wv    = (const float*)d_in[5];
  const float* cq    = (const float*)d_in[6];
  const float* sq    = (const float*)d_in[7];
  const float* ck    = (const float*)d_in[8];
  const float* sk    = (const float*)d_in[9];
  const float* wp    = (const float*)d_in[10];
  const float* tp    = (const float*)d_in[11];

  const size_t rows = (size_t)B_ * S_;          // 16384
  bf16* Qa = (bf16*)d_ws;                       // rows x 96
  bf16* Ka = Qa + rows * DAUG;                  // rows x 96
  bf16* Vt = Ka + rows * DAUG;                  // B x 64 x S (transposed V)

  proj_kernel<<<256, 128, 0, stream>>>(query, Wq, Qa, 0);
  proj_kernel<<<256, 128, 0, stream>>>(key,   Wk, Ka, 0);
  proj_kernel<<<256, 128, 0, stream>>>(value, Wv, Vt, 1);
  memb_kernel<<<128, 256, 0, stream>>>(Qa, Ka, cq, sq, ck, sk, wp);
  attn_kernel<<<256, 128, 0, stream>>>(Qa, Ka, Vt, (float*)d_out, wp, tp);
}
